// MoEBlock_75634374083060
// MI455X (gfx1250) — matrix operand order repaired and checked by reading, not compiled, not run on hardware
//
#include <hip/hip_runtime.h>
#include <hip/hip_bf16.h>
#include <math.h>

// Problem constants (match reference)
#define BB   2
#define SS   2048
#define DD   2048
#define FFF  8192
#define EE   16
#define TT   (BB*SS)      // 4096 tokens
#define CAP  640          // ceil(1.25 * 2 * 4096 / 16)

typedef __attribute__((ext_vector_type(16))) __bf16 bf16x16;
typedef __attribute__((ext_vector_type(8)))  float  f32x8;
typedef __attribute__((ext_vector_type(4)))  int    v4i;

// CDNA5 async global->LDS path (ASYNCcnt-tracked), with safe fallback
#if defined(__gfx1250__) && __has_builtin(__builtin_amdgcn_global_load_async_to_lds_b128)
#define HAVE_ASYNC_LDS 1
typedef __attribute__((address_space(1))) v4i gv4i;   // global
typedef __attribute__((address_space(3))) v4i lv4i;   // LDS
#else
#define HAVE_ASYNC_LDS 0
#endif

#if defined(__gfx1250__)
#if __has_builtin(__builtin_amdgcn_s_wait_asynccnt)
#define WAIT_ASYNC() __builtin_amdgcn_s_wait_asynccnt(0)
#else
#define WAIT_ASYNC() asm volatile("s_wait_asynccnt 0x0" ::: "memory")
#endif
#else
#define WAIT_ASYNC()
#endif

__device__ __forceinline__ unsigned short f2bf(float f) {
  unsigned int u = __builtin_bit_cast(unsigned int, f);
  u += 0x7FFFu + ((u >> 16) & 1u);           // round-to-nearest-even
  return (unsigned short)(u >> 16);
}

// ---------------- top-2 gate ----------------
__global__ void gate_kernel(const float* __restrict__ x, const float* __restrict__ Wg,
                            int* __restrict__ top_i, float* __restrict__ gates) {
  const int t   = blockIdx.x;
  const int tid = threadIdx.x;
  float partial[EE];
#pragma unroll
  for (int e = 0; e < EE; ++e) partial[e] = 0.f;
  for (int i = 0; i < DD / 256; ++i) {
    int d = tid + i * 256;
    float xv = x[(size_t)t * DD + d];
#pragma unroll
    for (int e = 0; e < EE; ++e) partial[e] += xv * Wg[d * EE + e];
  }
  __shared__ float red[EE];
  if (tid < EE) red[tid] = 0.f;
  __syncthreads();
  const int lane = tid & 31;
#pragma unroll
  for (int e = 0; e < EE; ++e) {
    float v = partial[e];
    for (int off = 16; off > 0; off >>= 1) v += __shfl_xor(v, off, 32);
    if (lane == 0) atomicAdd(&red[e], v);
  }
  __syncthreads();
  if (tid == 0) {
    float mx = -1e30f;
#pragma unroll
    for (int e = 0; e < EE; ++e) mx = fmaxf(mx, red[e]);
    float p[EE], sum = 0.f;
#pragma unroll
    for (int e = 0; e < EE; ++e) { p[e] = expf(red[e] - mx); sum += p[e]; }
    float inv = 1.f / sum;
    int i0 = 0; float v0 = -1.f;
#pragma unroll
    for (int e = 0; e < EE; ++e) { float pe = p[e] * inv; if (pe > v0) { v0 = pe; i0 = e; } }
    int i1 = 0; float v1 = -1.f;
#pragma unroll
    for (int e = 0; e < EE; ++e) {
      if (e == i0) continue;
      float pe = p[e] * inv; if (pe > v1) { v1 = pe; i1 = e; }
    }
    float s = v0 + v1;
    top_i[t * 2 + 0] = i0; top_i[t * 2 + 1] = i1;
    gates[t * 2 + 0] = v0 / s; gates[t * 2 + 1] = v1 / s;
  }
}

// ---------------- capacity scan (serial, slot-0 priority) ----------------
__global__ void scan_kernel(const int* __restrict__ top_i, float* __restrict__ gates,
                            int* __restrict__ comb_c, int* __restrict__ slot_tok) {
  if (threadIdx.x != 0 || blockIdx.x != 0) return;
  for (int s = 0; s < EE * CAP; ++s) slot_tok[s] = -1;
  int cnt0[EE], cnt1[EE];
  for (int e = 0; e < EE; ++e) { cnt0[e] = 0; cnt1[e] = 0; }
  for (int t = 0; t < TT; ++t) {               // slot-0 assignments first
    int e = top_i[t * 2];
    int p = cnt0[e]++;
    if (p < CAP) { slot_tok[e * CAP + p] = t; comb_c[t * 2] = p; }
    else         { gates[t * 2] = 0.f;         comb_c[t * 2] = CAP - 1; }
  }
  for (int t = 0; t < TT; ++t) {               // slot-1 offset by total slot-0 counts
    int e = top_i[t * 2 + 1];
    int p = cnt1[e]++ + cnt0[e];
    if (p < CAP) { slot_tok[e * CAP + p] = t; comb_c[t * 2 + 1] = p; }
    else         { gates[t * 2 + 1] = 0.f;     comb_c[t * 2 + 1] = CAP - 1; }
  }
}

// ---------------- dispatch: x -> Xe (bf16), zero empty slots ----------------
__global__ void dispatch_kernel(const float* __restrict__ x, const int* __restrict__ slot_tok,
                                unsigned short* __restrict__ Xe) {
  const int slot = blockIdx.x;                 // e*CAP + c
  const int t = slot_tok[slot];
  unsigned short* dst = Xe + (size_t)slot * DD;
  if (t >= 0) {
    const float* src = x + (size_t)t * DD;
    for (int d = threadIdx.x; d < DD; d += blockDim.x) dst[d] = f2bf(src[d]);
  } else {
    for (int d = threadIdx.x; d < DD; d += blockDim.x) dst[d] = 0;
  }
}

// ---------------- grouped GEMM ----------------
// Block tile 64(M) x 256(N), K-step 32, double-buffered LDS.
// 8 waves, wave tile 32x64: 8 WMMAs per K-step per wave, A frags reused 4x, B frags 2x.
// A tile staged via GLOBAL_LOAD_ASYNC_TO_LDS_B128 (ASYNCcnt); B tile f32->bf16 via
// register prefetch + VALU cvt (co-executes with XDL WMMA of the current tile).
template<int KTOT, int NTOT, bool DO_GELU, bool OUT_BF16>
__global__ __launch_bounds__(256)
void moe_gemm_kernel(const unsigned short* __restrict__ A,   // [E][CAP][KTOT] bf16
                     const float* __restrict__ Bw,           // [E][KTOT][NTOT] f32
                     const float* __restrict__ bias,         // [E][NTOT]
                     unsigned short* __restrict__ outB,
                     float* __restrict__ outF) {
  __shared__ __align__(16) unsigned short As[2][64][48];     // 64 x 32K (pad->48)
  __shared__ __align__(16) unsigned short BsT[2][256][48];   // transposed: 256N x 32K

  const int e   = blockIdx.z;
  const int m0  = blockIdx.y * 64;
  const int n0  = blockIdx.x * 256;
  const int tid = threadIdx.x;
  const int lane = tid & 31;
  const int w    = tid >> 5;
  const int wm   = w & 1;                 // 2 M groups of 32
  const int wn   = w >> 1;                // 4 N groups of 64

  const unsigned short* Ae = A + ((size_t)e * CAP + m0) * KTOT;
  const float* Be = Bw + (size_t)e * KTOT * NTOT + n0;

  // cooperative-load coordinates
  const int ar = tid >> 2;                // A: one uint4 (8 bf16) per thread
  const int ac = (tid & 3) * 8;
  const int br = tid >> 3;                // B: 32 f32 per thread, single K-row
  const int bc = (tid & 7) * 32;

  float breg[32];

  auto loadA = [&](int k0, int buf) {
#if HAVE_ASYNC_LDS
    void* gp = (void*)(Ae + (size_t)ar * KTOT + k0 + ac);
    void* lp = (void*)&As[buf][ar][ac];
    __builtin_amdgcn_global_load_async_to_lds_b128((gv4i*)gp, (lv4i*)lp, 0, 0);
#else
    *reinterpret_cast<uint4*>(&As[buf][ar][ac]) =
        *reinterpret_cast<const uint4*>(Ae + (size_t)ar * KTOT + k0 + ac);
#endif
  };
  auto loadBregs = [&](int k0) {
    const float* gp = Be + (size_t)(k0 + br) * NTOT + bc;
#pragma unroll
    for (int j = 0; j < 8; ++j) {
      float4 f = reinterpret_cast<const float4*>(gp)[j];
      breg[j * 4 + 0] = f.x; breg[j * 4 + 1] = f.y;
      breg[j * 4 + 2] = f.z; breg[j * 4 + 3] = f.w;
    }
  };
  auto storeB = [&](int buf) {
#pragma unroll
    for (int j = 0; j < 32; ++j) BsT[buf][bc + j][br] = f2bf(breg[j]);
  };

  // fragment coordinates (ISA 7.12.2: K split on lane bit 4)
  const int akb = (lane >> 4) * 8;
  const int bkb = (lane >> 4) * 8;
  const int al  = lane & 15;

  f32x8 acc[2][4] = {};

  // prologue: stage tile 0
  loadA(0, 0);
  loadBregs(0);
  storeB(0);
  WAIT_ASYNC();
  __syncthreads();

  constexpr int NSTEP = KTOT / 32;
  for (int s = 0; s < NSTEP; ++s) {
    const int cur = s & 1, nxt = cur ^ 1;
    if (s + 1 < NSTEP) {                       // prefetch next tile
      loadA((s + 1) * 32, nxt);
      loadBregs((s + 1) * 32);
    }

    union { uint4 u[2]; bf16x16 v; } af[2], bfr[4];
#pragma unroll
    for (int mt = 0; mt < 2; ++mt) {
      const int arow = wm * 32 + mt * 16 + al;
      af[mt].u[0] = *reinterpret_cast<const uint4*>(&As[cur][arow][akb]);
      af[mt].u[1] = *reinterpret_cast<const uint4*>(&As[cur][arow][akb + 16]);
    }
#pragma unroll
    for (int nt = 0; nt < 4; ++nt) {
      const int col = wn * 64 + nt * 16 + al;
      bfr[nt].u[0] = *reinterpret_cast<const uint4*>(&BsT[cur][col][bkb]);
      bfr[nt].u[1] = *reinterpret_cast<const uint4*>(&BsT[cur][col][bkb + 16]);
    }
#pragma unroll
    for (int nt = 0; nt < 4; ++nt)
#pragma unroll
      for (int mt = 0; mt < 2; ++mt)
        acc[mt][nt] = __builtin_amdgcn_wmma_f32_16x16x32_bf16(
            false, af[mt].v, false, bfr[nt].v, (short)0, acc[mt][nt], false, false);

    if (s + 1 < NSTEP) {
      storeB(nxt);                             // cvt+store overlaps XDL WMMA
      WAIT_ASYNC();
    }
    __syncthreads();
  }

  // epilogue: C/D layout — lanes 0-15: M=vgpr, lanes 16-31: M=vgpr+8; N=lane&15
  const int mb = m0 + wm * 32 + 8 * (lane >> 4);
  const int nb = n0 + wn * 64 + al;
#pragma unroll
  for (int mt = 0; mt < 2; ++mt) {
#pragma unroll
    for (int nt = 0; nt < 4; ++nt) {
      const int col = nb + nt * 16;
      const float bv = bias[(size_t)e * NTOT + col];
#pragma unroll
      for (int i = 0; i < 8; ++i) {
        float v = acc[mt][nt][i] + bv;
        if (DO_GELU) v = 0.5f * v * (1.f + erff(v * 0.70710678118f));  // exact GELU
        const size_t o = ((size_t)e * CAP + (mb + mt * 16 + i)) * NTOT + col;
        if (OUT_BF16) outB[o] = f2bf(v); else outF[o] = v;
      }
    }
  }
}

// ---------------- combine ----------------
__global__ void combine_kernel(const float* __restrict__ Ye, const int* __restrict__ top_i,
                               const int* __restrict__ comb_c, const float* __restrict__ gates,
                               float* __restrict__ y) {
  const int t = blockIdx.x;
  const int e0 = top_i[t * 2], e1 = top_i[t * 2 + 1];
  const int c0 = comb_c[t * 2], c1 = comb_c[t * 2 + 1];
  const float g0 = gates[t * 2], g1 = gates[t * 2 + 1];
  const float* y0 = Ye + ((size_t)e0 * CAP + c0) * DD;
  const float* y1 = Ye + ((size_t)e1 * CAP + c1) * DD;
  for (int d = threadIdx.x; d < DD; d += blockDim.x)
    y[(size_t)t * DD + d] = g0 * y0[d] + g1 * y1[d];
}

extern "C" void kernel_launch(void* const* d_in, const int* in_sizes, int n_in,
                              void* d_out, int out_size, void* d_ws, size_t ws_size,
                              hipStream_t stream) {
  const float* x  = (const float*)d_in[0];
  const float* Wg = (const float*)d_in[1];
  const float* W1 = (const float*)d_in[2];
  const float* b1 = (const float*)d_in[3];
  const float* W2 = (const float*)d_in[4];
  const float* b2 = (const float*)d_in[5];
  float* y = (float*)d_out;

  char* base = (char*)d_ws;
  size_t off = 0;
  auto alloc = [&](size_t bytes) -> char* {
    char* p = base + off;
    off = (off + bytes + 255) & ~(size_t)255;
    return p;
  };
  int*   top_i   = (int*)  alloc(sizeof(int)   * (size_t)TT * 2);
  float* gates   = (float*)alloc(sizeof(float) * (size_t)TT * 2);
  int*   comb_c  = (int*)  alloc(sizeof(int)   * (size_t)TT * 2);
  int*   slot_tk = (int*)  alloc(sizeof(int)   * (size_t)EE * CAP);
  unsigned short* Xe = (unsigned short*)alloc((size_t)2 * EE * CAP * DD);   //  42 MB
  unsigned short* H  = (unsigned short*)alloc((size_t)2 * EE * CAP * FFF);  // 168 MB
  float*          Ye = (float*)alloc(sizeof(float) * (size_t)EE * CAP * DD);//  84 MB

  gate_kernel<<<TT, 256, 0, stream>>>(x, Wg, top_i, gates);
  scan_kernel<<<1, 32, 0, stream>>>(top_i, gates, comb_c, slot_tk);
  dispatch_kernel<<<EE * CAP, 256, 0, stream>>>(x, slot_tk, Xe);
  moe_gemm_kernel<DD,  FFF, true,  true ><<<dim3(FFF / 256, CAP / 64, EE), 256, 0, stream>>>(
      Xe, W1, b1, H, nullptr);
  moe_gemm_kernel<FFF, DD,  false, false><<<dim3(DD  / 256, CAP / 64, EE), 256, 0, stream>>>(
      H, W2, b2, nullptr, Ye);
  combine_kernel<<<TT, 256, 0, stream>>>(Ye, top_i, comb_c, gates, y);
}
